// MHA_10703058502125
// MI455X (gfx1250) — matrix operand order repaired and checked by reading, not compiled, not run on hardware
//
#include <hip/hip_runtime.h>
#include <hip/hip_bf16.h>

// MHA with per-head projections, flash-attention style, bf16 WMMA on gfx1250.
// B=4 SQ=SK=2048 E=1024 H=16 HD=64. Compute-bound -> v_wmma_f32_16x16x32_bf16.
// Round 3: fix tensor_load_to_lds arity (this toolchain uses the 6-arg form).

#define B_   4
#define SQ_  2048
#define SK_  2048
#define E_   1024
#define H_   16
#define HD_  64

typedef __attribute__((ext_vector_type(16))) __bf16 v16bf;
typedef __attribute__((ext_vector_type(8)))  __bf16 v8bf;
typedef __attribute__((ext_vector_type(8)))  float  v8f;
typedef __attribute__((ext_vector_type(4)))  float  v4f;

#if defined(__gfx1250__) && __has_builtin(__builtin_amdgcn_tensor_load_to_lds) && \
    __has_builtin(__builtin_amdgcn_s_wait_tensorcnt)
#define USE_TDM 1
#else
#define USE_TDM 0
#endif

static __device__ __forceinline__ v8f bwmma(v16bf a, v16bf b, v8f c) {
  // D = A(16x32 bf16) * B(32x16 bf16) + C(16x16 f32)
  return __builtin_amdgcn_wmma_f32_16x16x32_bf16(
      /*neg_a=*/false, a, /*neg_b=*/false, b,
      /*c_mod=*/(short)0, c, /*reuse_a=*/false, /*reuse_b=*/false);
}

// B-matrix fragment: lane c holds column n=c%16; lane<16: elems 0..7 = K0..7,
// 8..15 = K16..23; lane>=16: K8..15 / K24..31 (caller applies +8 via blo).
static __device__ __forceinline__ v16bf load_b_frag(const __bf16* p) {
  v8bf lo = *reinterpret_cast<const v8bf*>(p);
  v8bf hi = *reinterpret_cast<const v8bf*>(p + 16);
  v16bf r;
#pragma unroll
  for (int i = 0; i < 8; ++i) { r[i] = lo[i]; r[i + 8] = hi[i]; }
  return r;
}

// A-matrix fragment (16-bit A 16x32 layout): lane<16: elems 0..7 = K0..7,
// 8..15 = K16..23; lane>=16: K8..15 / K24..31. p is pre-offset by alo (0|8).
static __device__ __forceinline__ v16bf load_a_frag(const __bf16* p) {
  v8bf lo = *reinterpret_cast<const v8bf*>(p);
  v8bf hi = *reinterpret_cast<const v8bf*>(p + 16);
  v16bf r;
#pragma unroll
  for (int i = 0; i < 8; ++i) { r[i] = lo[i]; r[i + 8] = hi[i]; }
  return r;
}

// A-fragment from fp32 row (converts to bf16 in-register). p pre-offset by alo.
static __device__ __forceinline__ v16bf make_a_from_f32(const float* p) {
  v4f x0 = *reinterpret_cast<const v4f*>(p);
  v4f x1 = *reinterpret_cast<const v4f*>(p + 4);
  v4f x2 = *reinterpret_cast<const v4f*>(p + 16);
  v4f x3 = *reinterpret_cast<const v4f*>(p + 20);
  v16bf a;
#pragma unroll
  for (int i = 0; i < 4; ++i) {
    a[i]      = (__bf16)x0[i];
    a[i + 4]  = (__bf16)x1[i];
    a[i + 8]  = (__bf16)x2[i];
    a[i + 12] = (__bf16)x3[i];
  }
  return a;
}

#if USE_TDM
typedef unsigned int v4u  __attribute__((ext_vector_type(4)));
typedef int          v8i_ __attribute__((ext_vector_type(8)));
typedef int          v4i_ __attribute__((ext_vector_type(4)));

// Issue one 2-D TDM tile load (bf16 elements) global -> LDS.
// tensor_d0/tile_d0 along the contiguous axis, stride0 in elements.
static __device__ __forceinline__ void tdm_load_2d(const __bf16* gsrc,
                                                   unsigned lds_off,
                                                   unsigned tensor_d0,
                                                   unsigned tensor_d1,
                                                   unsigned tile_d0,
                                                   unsigned tile_d1,
                                                   unsigned long long stride0) {
  unsigned long long ga = (unsigned long long)(uintptr_t)gsrc;
  v4u g0;
  g0[0] = 1u;                                   // count=1, user descriptor
  g0[1] = lds_off;                              // D#.lds_addr (bytes)
  g0[2] = (unsigned)ga;                         // global_addr[31:0]
  g0[3] = (unsigned)((ga >> 32) & 0x01FFFFFFu)  // global_addr[56:32]
          | (2u << 30);                         // type = 2 ("image")
  v8i_ g1;
  g1[0] = (int)(1u << 16);                      // data_size=1 -> 2 bytes/elem
  g1[1] = (int)((tensor_d0 & 0xFFFFu) << 16);   // tensor_dim0[15:0]
  g1[2] = (int)(((tensor_d0 >> 16) & 0xFFFFu) | ((tensor_d1 & 0xFFFFu) << 16));
  g1[3] = (int)(((tensor_d1 >> 16) & 0xFFFFu) | ((tile_d0 & 0xFFFFu) << 16));
  g1[4] = (int)(tile_d1 & 0xFFFFu);             // tile_dim1 (tile_dim2 = 0)
  g1[5] = (int)(stride0 & 0xFFFFFFFFull);       // tensor_dim0_stride[31:0]
  g1[6] = (int)((stride0 >> 32) & 0xFFFFull);   // [47:32]; dim1_stride = 0
  g1[7] = 0;
  v4i_ gz4 = {0, 0, 0, 0};
  v8i_ gz8 = {0, 0, 0, 0, 0, 0, 0, 0};
  // 6-arg form: (group0, group1, group2, group3, extra group, cpol)
  __builtin_amdgcn_tensor_load_to_lds(g0, g1, gz4, gz4, gz8, 0);
}
#endif

// ---------------- Kernel 1: weight fp32 [H,E,HD] -> bf16 transposed [H,HD,E]
__global__ __launch_bounds__(256) void wconv_kernel(const float* __restrict__ W,
                                                    __bf16* __restrict__ Wt) {
  int idx = blockIdx.x * 256 + threadIdx.x;     // over H*HD*E = 1M
  int e = idx % E_;
  int d = (idx / E_) % HD_;
  int h = idx / (E_ * HD_);
  Wt[((size_t)h * HD_ + d) * E_ + e] = (__bf16)W[((size_t)h * E_ + e) * HD_ + d];
}

// ---------------- Kernel 2: projection GEMM  Out[b,h,s,d] = X[b,s,:] . Wt[h,d,:]
// One wave computes a 32(s) x 64(d) tile (two 16-row groups share B-frags).
// transposed!=0 stores Out as [b,h,d,s] (V -> contiguous PV B-frags later).
__global__ __launch_bounds__(256) void proj_kernel(const float* __restrict__ X,
                                                   const __bf16* __restrict__ Wt,
                                                   __bf16* __restrict__ Out,
                                                   int transposed) {
  const int wave = threadIdx.x >> 5;
  const int lane = threadIdx.x & 31;
  const int tile = blockIdx.x * 8 + wave;
  const int MT = SQ_ / 32;                      // 64 row tiles
  const int mt = tile % MT;
  const int h  = (tile / MT) % H_;
  const int b  = tile / (MT * H_);
  const int m0 = mt * 32;
  const int c    = lane & 15;
  const int half = lane >> 4;
  const int alo  = half ? 8 : 0;
  const int blo  = half ? 8 : 0;

  const float*  xrow0 = X + ((size_t)b * SQ_ + (m0 + c)) * E_;
  const float*  xrow1 = xrow0 + (size_t)16 * E_;
  const __bf16* wbase = Wt + (size_t)h * HD_ * E_;

  v8f acc[2][4] = {};
  for (int e0 = 0; e0 < E_; e0 += 32) {
    v16bf a0 = make_a_from_f32(xrow0 + e0 + alo);
    v16bf a1 = make_a_from_f32(xrow1 + e0 + alo);
#pragma unroll
    for (int j = 0; j < 4; ++j) {
      v16bf w = load_b_frag(wbase + (size_t)(j * 16 + c) * E_ + e0 + blo);
      acc[0][j] = bwmma(a0, w, acc[0][j]);
      acc[1][j] = bwmma(a1, w, acc[1][j]);
    }
  }
#pragma unroll
  for (int g = 0; g < 2; ++g) {
#pragma unroll
    for (int j = 0; j < 4; ++j) {
#pragma unroll
      for (int r = 0; r < 8; ++r) {
        const int M = g * 16 + r + half * 8;
        const int n = j * 16 + c;
        size_t idx;
        if (transposed)
          idx = (((size_t)(b * H_ + h) * HD_) + n) * (size_t)SK_ + (m0 + M);
        else
          idx = (((size_t)(b * H_ + h) * SQ_) + (m0 + M)) * (size_t)HD_ + n;
        Out[idx] = (__bf16)acc[g][j][r];
      }
    }
  }
}

// ---------------- Kernel 3: flash attention. One wave = 32 q-rows. The 8 waves
// of a block share (b,h); K(32x64) and V(64x32) tiles are staged into LDS per
// 32-key iteration (TDM double-buffered when available), then consumed as WMMA
// fragments via ds loads. fp32 online softmax; output interleaved [b,s,d*H+h].
__global__ __launch_bounds__(256) void attn_kernel(const __bf16* __restrict__ Qb,
                                                   const __bf16* __restrict__ Kb,
                                                   const __bf16* __restrict__ Vt,
                                                   float* __restrict__ Out) {
  __shared__ __align__(16) __bf16 kvbuf[2][4096];     // [buf][Ktile 2048 | Vtile 2048]
  __shared__ __align__(16) __bf16 ptile[8][2][512];   // per-wave, per-rowgroup P
  const int wave = threadIdx.x >> 5;
  const int lane = threadIdx.x & 31;
  const int tile = blockIdx.x * 8 + wave;
  const int MT = SQ_ / 32;                      // 64 q tiles per (b,h)
  const int qt = tile % MT;
  const int h  = (tile / MT) % H_;
  const int b  = tile / (MT * H_);
  const int q0 = qt * 32;
  const int c    = lane & 15;
  const int half = lane >> 4;
  const int alo  = half ? 8 : 0;
  const int blo  = half ? 8 : 0;

  const __bf16* qbase = Qb + ((size_t)(b * H_ + h) * SQ_) * HD_;
  const __bf16* kbase = Kb + ((size_t)(b * H_ + h) * SQ_) * HD_;
  const __bf16* vbase = Vt + ((size_t)(b * H_ + h) * HD_) * SK_;

  // Q A-fragments for two row groups x two 32-wide K chunks; held in registers.
  v16bf qa[2][2];
#pragma unroll
  for (int g = 0; g < 2; ++g) {
    const __bf16* qr = qbase + (size_t)(q0 + g * 16 + c) * HD_;
    qa[g][0] = load_a_frag(qr + 0  + alo);
    qa[g][1] = load_a_frag(qr + 32 + alo);
  }

  const float cl = 0.125f * 1.44269504f;        // 1/sqrt(64) * log2(e)
  float m[2][8], l[2][8];
  v8f o[2][4] = {};
#pragma unroll
  for (int g = 0; g < 2; ++g)
#pragma unroll
    for (int r = 0; r < 8; ++r) { m[g][r] = -1e30f; l[g][r] = 0.0f; }

  const int NIT = SK_ / 32;                     // 64 key iterations

#if USE_TDM
  if (wave == 0) {
    tdm_load_2d(kbase, (unsigned)(uintptr_t)&kvbuf[0][0],
                HD_, SQ_, HD_, 32, HD_);                    // K tile rows=keys
    tdm_load_2d(vbase, (unsigned)(uintptr_t)&kvbuf[0][2048],
                SK_, HD_, 32, HD_, SK_);                    // V tile rows=d
  }
#endif

  for (int it = 0; it < NIT; ++it) {
    const int k0 = it * 32;
#if USE_TDM
    const int cur = it & 1;
    if (wave == 0) {
      if (it + 1 < NIT) {
        const int k1 = k0 + 32;
        tdm_load_2d(kbase + (size_t)k1 * HD_,
                    (unsigned)(uintptr_t)&kvbuf[cur ^ 1][0],
                    HD_, SQ_, HD_, 32, HD_);
        tdm_load_2d(vbase + k1,
                    (unsigned)(uintptr_t)&kvbuf[cur ^ 1][2048],
                    SK_, HD_, 32, HD_, SK_);
        __builtin_amdgcn_s_wait_tensorcnt(2);   // current pair complete
      } else {
        __builtin_amdgcn_s_wait_tensorcnt(0);
      }
    }
    __syncthreads();
#else
    const int cur = 0;
    __syncthreads();                            // prior reads done before overwrite
    {
      const int i = threadIdx.x * 8;            // 256 thr x 8 elems = 2048
      *reinterpret_cast<v8bf*>(&kvbuf[0][i]) =
          *reinterpret_cast<const v8bf*>(kbase + (size_t)(k0 + (i >> 6)) * HD_ + (i & 63));
      *reinterpret_cast<v8bf*>(&kvbuf[0][2048 + i]) =
          *reinterpret_cast<const v8bf*>(vbase + (size_t)(i >> 5) * SK_ + k0 + (i & 31));
    }
    __syncthreads();
#endif

    const __bf16* kt = &kvbuf[cur][0];          // [key 0..31][d 0..63]
    const __bf16* vt = &kvbuf[cur][2048];       // [d 0..63][key 0..31]

#pragma unroll
    for (int g = 0; g < 2; ++g) {
      // ---- scores: two 16-key blocks, K over HD in two chunks (from LDS)
      v8f s0 = {}, s1 = {};
      s0 = bwmma(qa[g][0], load_b_frag(kt + (size_t)c * HD_ + 0  + blo), s0);
      s0 = bwmma(qa[g][1], load_b_frag(kt + (size_t)c * HD_ + 32 + blo), s0);
      s1 = bwmma(qa[g][0], load_b_frag(kt + (size_t)(16 + c) * HD_ + 0  + blo), s1);
      s1 = bwmma(qa[g][1], load_b_frag(kt + (size_t)(16 + c) * HD_ + 32 + blo), s1);

      // ---- online softmax per q-row (row M=r+8*half lives in one lane half)
      __bf16* sp = &ptile[wave][g][0];
#pragma unroll
      for (int r = 0; r < 8; ++r) {
        float t = fmaxf(s0[r], s1[r]);
        t = fmaxf(t, __shfl_xor(t, 1));
        t = fmaxf(t, __shfl_xor(t, 2));
        t = fmaxf(t, __shfl_xor(t, 4));
        t = fmaxf(t, __shfl_xor(t, 8));
        const float mn    = fmaxf(m[g][r], t);
        const float alpha = __builtin_exp2f((m[g][r] - mn) * cl);
        const float p0    = __builtin_exp2f((s0[r] - mn) * cl);
        const float p1    = __builtin_exp2f((s1[r] - mn) * cl);
        float rs = p0 + p1;
        rs += __shfl_xor(rs, 1);
        rs += __shfl_xor(rs, 2);
        rs += __shfl_xor(rs, 4);
        rs += __shfl_xor(rs, 8);
        l[g][r] = l[g][r] * alpha + rs;
        m[g][r] = mn;
#pragma unroll
        for (int j = 0; j < 4; ++j) o[g][j][r] *= alpha;
        const int row = r + half * 8;
        sp[row * 32 + c]      = (__bf16)p0;
        sp[row * 32 + 16 + c] = (__bf16)p1;
      }

      // ---- P (16x32) as A fragment via per-wave LDS relayout
      const v16bf pa = load_a_frag(sp + c * 32 + alo);

      // ---- PV: B columns n=d are contiguous rows of the LDS V tile
#pragma unroll
      for (int j = 0; j < 4; ++j) {
        const v16bf vb = load_b_frag(vt + (size_t)(j * 16 + c) * 32 + blo);
        o[g][j] = bwmma(pa, vb, o[g][j]);
      }
    }
    __syncthreads();                            // all reads done before restage
  }

  // ---- epilogue: out[b, s, d*H + h] fp32
#pragma unroll
  for (int g = 0; g < 2; ++g) {
#pragma unroll
    for (int r = 0; r < 8; ++r) {
      const float inv = 1.0f / l[g][r];
      const int s = q0 + g * 16 + r + half * 8;
#pragma unroll
      for (int j = 0; j < 4; ++j) {
        const int d = j * 16 + c;
        Out[((size_t)b * SQ_ + s) * (size_t)(H_ * HD_) + (size_t)d * H_ + h] =
            o[g][j][r] * inv;
      }
    }
  }
}

extern "C" void kernel_launch(void* const* d_in, const int* in_sizes, int n_in,
                              void* d_out, int out_size, void* d_ws, size_t ws_size,
                              hipStream_t stream) {
  const float* q_in = (const float*)d_in[0];
  const float* k_in = (const float*)d_in[1];
  const float* v_in = (const float*)d_in[2];
  const float* Wq   = (const float*)d_in[3];
  const float* Wk   = (const float*)d_in[4];
  const float* Wv   = (const float*)d_in[5];
  float* out = (float*)d_out;

  // Workspace (bf16): Wqt/Wkt/Wvt [H,HD,E] (2MB ea), Qb/Kb [B,H,SQ,HD] (16MB ea),
  // Vt [B,H,HD,SK] (16MB). Total ~54MB.
  const size_t wElems = (size_t)H_ * E_ * HD_;        // 1,048,576
  const size_t pElems = (size_t)B_ * H_ * SQ_ * HD_;  // 8,388,608
  __bf16* Wqt = (__bf16*)d_ws;
  __bf16* Wkt = Wqt + wElems;
  __bf16* Wvt = Wkt + wElems;
  __bf16* Qb  = Wvt + wElems;
  __bf16* Kb  = Qb + pElems;
  __bf16* Vt  = Kb + pElems;

  const int wblocks = (int)(wElems / 256);            // 4096
  wconv_kernel<<<wblocks, 256, 0, stream>>>(Wq, Wqt);
  wconv_kernel<<<wblocks, 256, 0, stream>>>(Wk, Wkt);
  wconv_kernel<<<wblocks, 256, 0, stream>>>(Wv, Wvt);

  const int tiles   = B_ * H_ * (SQ_ / 32);           // 4096 waves
  const int pblocks = tiles / 8;                      // 512 blocks x 256 thr
  proj_kernel<<<pblocks, 256, 0, stream>>>(q_in, Wqt, Qb, 0);
  proj_kernel<<<pblocks, 256, 0, stream>>>(k_in, Wkt, Kb, 0);
  proj_kernel<<<pblocks, 256, 0, stream>>>(v_in, Wvt, Vt, 1);

  attn_kernel<<<pblocks, 256, 0, stream>>>(Qb, Kb, Vt, out);
}
